// Seq2SeqAE_77506979824199
// MI455X (gfx1250) — matrix operand order repaired and checked by reading, not compiled, not run on hardware
//
#include <hip/hip_runtime.h>
#include <hip/hip_bf16.h>
#include <math.h>

#define VOCAB 32000
#define EMB 256
#define HID 512
#define LAT 128
#define BB 16
#define TT 128

typedef __attribute__((ext_vector_type(16))) __bf16 bf16x16;
typedef __attribute__((ext_vector_type(8)))  float  floatx8;
typedef __attribute__((ext_vector_type(4))) unsigned int uint4v;
typedef __attribute__((ext_vector_type(8))) int int8v;
typedef __attribute__((ext_vector_type(4))) int int4v;

// explicit global-address-space pointer (keeps global_load codegen after the
// anti-LICM pointer laundering below)
typedef const __bf16 __attribute__((address_space(1)))* gbf16p;

__device__ __forceinline__ floatx8 wmma_bf16(bf16x16 a, bf16x16 b, floatx8 c) {
  return __builtin_amdgcn_wmma_f32_16x16x32_bf16(false, a, false, b, (short)0, c, false, false);
}

// A-fragment (16x32 bf16, M x K), row-major activations, ld = K stride.
// Lanes 0-15: row M=lane, K = kb+0..15; lanes 16-31: same rows, K = kb+16..31.
template <typename P>
__device__ __forceinline__ bf16x16 load_a_frag(P base, int ld, int kb, int lane) {
  const int r = lane & 15, h = (lane >> 4) & 1;
  P p = base + (size_t)r * ld + kb + h * 16;
  bf16x16 f;
#pragma unroll
  for (int j = 0; j < 16; ++j) f[j] = p[j];
  return f;
}

// B-fragment (32x16 bf16, K x N) where B[k][n] = W[row0+n][k], W row-major with ld = K stride.
// Lanes 0-15 hold K = kb+0..15 for column n=lane; lanes 16-31 hold K = kb+16..31.
template <typename P>
__device__ __forceinline__ bf16x16 load_b_frag(P base, int ld, int row0, int kb, int lane) {
  const int r = lane & 15, h = (lane >> 4) & 1;
  P p = base + (size_t)(row0 + r) * ld + kb + h * 16;
  bf16x16 f;
#pragma unroll
  for (int j = 0; j < 16; ++j) f[j] = p[j];
  return f;
}

__device__ __forceinline__ float sigmoidf_(float x) { return 1.0f / (1.0f + __expf(-x)); }

// ---------------------------------------------------------------- prep kernels
__global__ void f32_to_bf16_kernel(const float* __restrict__ src, __bf16* __restrict__ dst, int n) {
  int i = blockIdx.x * blockDim.x + threadIdx.x;
  if (i < n) dst[i] = (__bf16)src[i];
}

__global__ void gather_embed_kernel(const int* __restrict__ x, const float* __restrict__ emb,
                                    __bf16* __restrict__ e_seq) {
  int i = blockIdx.x * blockDim.x + threadIdx.x;
  if (i >= TT * BB * EMB) return;
  int t = i / (BB * EMB);
  int rem = i % (BB * EMB);
  int b = rem / EMB, k = rem % EMB;
  e_seq[i] = (__bf16)emb[(size_t)x[b * TT + t] * EMB + k];
}

__global__ void zero_t0_kernel(float* __restrict__ out) {
  int i = blockIdx.x * blockDim.x + threadIdx.x;
  if (i >= BB * VOCAB) return;
  int b = i / VOCAB, v = i % VOCAB;
  out[(size_t)b * TT * VOCAB + v] = 0.0f;
}

__global__ void init_tok_kernel(const int* __restrict__ x, int* __restrict__ tok) {
  int b = threadIdx.x;
  if (b < BB) tok[b] = x[b * TT];
}

// ---------------------------------------------------------------- encoder GRU
// One block, 32 waves; wave w owns hidden columns [16w, 16w+16). 128 sequential steps.
__global__ __launch_bounds__(1024) void enc_gru_kernel(
    const __bf16* __restrict__ e_seq, const __bf16* __restrict__ Wih,
    const __bf16* __restrict__ Whh, const float* __restrict__ bih,
    const float* __restrict__ bhh, float* __restrict__ h_out) {
  __shared__ float  h_f[BB * HID];
  __shared__ __bf16 h_b[BB * HID];
  const int tid = threadIdx.x;
  const int lane = tid & 31;
  const int wv = tid >> 5;
  const int jw = wv * 16;
  const int r = lane & 15, half = (lane >> 4) & 1;
  const int col = jw + r;

  for (int i = tid; i < BB * HID; i += 1024) { h_f[i] = 0.0f; h_b[i] = (__bf16)0.0f; }

  const float b_r  = bih[col]           + bhh[col];
  const float b_z  = bih[HID + col]     + bhh[HID + col];
  const float b_in = bih[2 * HID + col];
  const float b_hn = bhh[2 * HID + col];
  __syncthreads();

  unsigned long long wih_u = (unsigned long long)(uintptr_t)Wih;
  unsigned long long whh_u = (unsigned long long)(uintptr_t)Whh;

  for (int t = 0; t < TT; ++t) {
    // Defeat LICM: keep the weight fragment loads inside the time loop so they
    // stream from (shared, L2-resident) global memory instead of being hoisted
    // and spilled to per-wave scratch. Cast back to address_space(1) so the
    // loads stay global_load (not flat_load, which also ties up DScnt).
    asm volatile("" : "+s"(wih_u), "+s"(whh_u));
    gbf16p Wih_t = (gbf16p)(uintptr_t)wih_u;
    gbf16p Whh_t = (gbf16p)(uintptr_t)whh_u;

    const __bf16* e_t = e_seq + (size_t)t * BB * EMB;
    floatx8 c_r, c_z, c_in, c_hn;
#pragma unroll
    for (int i = 0; i < 8; ++i) { c_r[i] = b_r; c_z[i] = b_z; c_in[i] = b_in; c_hn[i] = b_hn; }

    for (int kb = 0; kb < EMB; kb += 32) {
      bf16x16 a = load_a_frag(e_t, EMB, kb, lane);
      c_r  = wmma_bf16(a, load_b_frag(Wih_t, EMB, jw,           kb, lane), c_r);
      c_z  = wmma_bf16(a, load_b_frag(Wih_t, EMB, HID + jw,     kb, lane), c_z);
      c_in = wmma_bf16(a, load_b_frag(Wih_t, EMB, 2 * HID + jw, kb, lane), c_in);
    }
    for (int kb = 0; kb < HID; kb += 32) {
      bf16x16 a = load_a_frag(h_b, HID, kb, lane);
      c_r  = wmma_bf16(a, load_b_frag(Whh_t, HID, jw,           kb, lane), c_r);
      c_z  = wmma_bf16(a, load_b_frag(Whh_t, HID, HID + jw,     kb, lane), c_z);
      c_hn = wmma_bf16(a, load_b_frag(Whh_t, HID, 2 * HID + jw, kb, lane), c_hn);
    }
    __syncthreads();  // all reads of h_b done before anyone overwrites it
#pragma unroll
    for (int i = 0; i < 8; ++i) {
      const int row = i + 8 * half;                  // C/D layout: VGPR i -> M = i + 8*half
      const float rg = sigmoidf_(c_r[i]);
      const float zg = sigmoidf_(c_z[i]);
      const float ng = tanhf(c_in[i] + rg * c_hn[i]);
      const float hv = (1.0f - zg) * ng + zg * h_f[row * HID + col];
      h_f[row * HID + col] = hv;
      h_b[row * HID + col] = (__bf16)hv;
    }
    __syncthreads();  // writes visible before next step's reads
  }
  for (int i = tid; i < BB * HID; i += 1024) h_out[i] = h_f[i];
}

// ---------------------------------------------------------------- latent FCs (tiny, scalar f32)
__global__ void fc_enc_kernel(const float* __restrict__ h, const float* __restrict__ W,
                              const float* __restrict__ b, float* __restrict__ zbuf,
                              float* __restrict__ out_z) {
  const int idx = blockIdx.x * blockDim.x + threadIdx.x;
  if (idx >= BB * LAT) return;
  const int bb = idx / LAT, l = idx % LAT;
  float s = b[l];
  for (int k = 0; k < HID; ++k) s += h[bb * HID + k] * W[l * HID + k];
  zbuf[idx] = s;
  out_z[idx] = s;
}

__global__ void fc_dec_kernel(const float* __restrict__ z, const float* __restrict__ W,
                              const float* __restrict__ b, float* __restrict__ h_f,
                              __bf16* __restrict__ h_b) {
  const int idx = blockIdx.x * blockDim.x + threadIdx.x;
  if (idx >= BB * HID) return;
  const int bb = idx / HID, j = idx % HID;
  float s = b[j];
  for (int l = 0; l < LAT; ++l) s += z[bb * LAT + l] * W[j * LAT + l];
  const float hv = tanhf(s);
  h_f[idx] = hv;
  h_b[idx] = (__bf16)hv;
}

// ---------------------------------------------------------------- decoder GRU (one step)
__global__ __launch_bounds__(1024) void dec_gru_kernel(
    const int* __restrict__ tok, const float* __restrict__ emb_dec,
    const __bf16* __restrict__ Wih, const __bf16* __restrict__ Whh,
    const float* __restrict__ bih, const float* __restrict__ bhh,
    float* __restrict__ h_f, __bf16* __restrict__ h_b) {
  __shared__ __bf16 e_s[BB * EMB];
  const int tid = threadIdx.x;
  const int lane = tid & 31;
  const int wv = tid >> 5;
  const int jw = wv * 16;
  const int r = lane & 15, half = (lane >> 4) & 1;
  const int col = jw + r;

  for (int i = tid; i < BB * EMB; i += 1024) {
    const int b = i / EMB, k = i % EMB;
    e_s[i] = (__bf16)emb_dec[(size_t)tok[b] * EMB + k];
  }
  __syncthreads();

  floatx8 c_r, c_z, c_in, c_hn;
  const float b_r  = bih[col]           + bhh[col];
  const float b_z  = bih[HID + col]     + bhh[HID + col];
  const float b_in = bih[2 * HID + col];
  const float b_hn = bhh[2 * HID + col];
#pragma unroll
  for (int i = 0; i < 8; ++i) { c_r[i] = b_r; c_z[i] = b_z; c_in[i] = b_in; c_hn[i] = b_hn; }

  for (int kb = 0; kb < EMB; kb += 32) {
    bf16x16 a = load_a_frag(e_s, EMB, kb, lane);
    c_r  = wmma_bf16(a, load_b_frag(Wih, EMB, jw,           kb, lane), c_r);
    c_z  = wmma_bf16(a, load_b_frag(Wih, EMB, HID + jw,     kb, lane), c_z);
    c_in = wmma_bf16(a, load_b_frag(Wih, EMB, 2 * HID + jw, kb, lane), c_in);
  }
  for (int kb = 0; kb < HID; kb += 32) {
    bf16x16 a = load_a_frag(h_b, HID, kb, lane);
    c_r  = wmma_bf16(a, load_b_frag(Whh, HID, jw,           kb, lane), c_r);
    c_z  = wmma_bf16(a, load_b_frag(Whh, HID, HID + jw,     kb, lane), c_z);
    c_hn = wmma_bf16(a, load_b_frag(Whh, HID, 2 * HID + jw, kb, lane), c_hn);
  }
  __syncthreads();  // all h_b reads complete before overwrite (single block)
#pragma unroll
  for (int i = 0; i < 8; ++i) {
    const int row = i + 8 * half;
    const float rg = sigmoidf_(c_r[i]);
    const float zg = sigmoidf_(c_z[i]);
    const float ng = tanhf(c_in[i] + rg * c_hn[i]);
    const float hv = (1.0f - zg) * ng + zg * h_f[row * HID + col];
    h_f[row * HID + col] = hv;
    h_b[row * HID + col] = (__bf16)hv;
  }
}

// ---------------------------------------------------------------- logits (16x32000) + partial argmax
// 250 blocks x 8 waves = 2000 tiles of 16 vocab columns; K=512 via 16 bf16 WMMAs per wave.
// h2 (16x512 bf16 = 16KB) is staged into LDS once per block by the Tensor Data Mover.
__global__ __launch_bounds__(256) void logits_kernel(
    const __bf16* __restrict__ h_b, const __bf16* __restrict__ Wout,
    const float* __restrict__ bout, float* __restrict__ out,
    float* __restrict__ pv, int* __restrict__ pi, int t) {
  __shared__ __bf16 h_s[BB * HID];   // 16 KB
  const int lane = threadIdx.x & 31;
  const int wv = threadIdx.x >> 5;
  const int tile = blockIdx.x * 8 + wv;
  const int v0 = tile * 16;
  const int r = lane & 15, half = (lane >> 4) & 1;

  if (wv == 0) {
    // Tensor DMA descriptor: 1 row of 8192 x 2B elements (contiguous 16 KB), global -> LDS.
    const unsigned long long ga = (unsigned long long)(uintptr_t)h_b;   // 57-bit global byte address
    const unsigned int lds_off = (unsigned int)(uintptr_t)&h_s[0];      // low 32 bits = LDS byte offset
    uint4v g0;
    g0[0] = 1u;                                        // count=1 (valid descriptor)
    g0[1] = lds_off;                                   // lds_addr
    g0[2] = (unsigned int)(ga & 0xffffffffu);          // global_addr[31:0] -> bits 95:64
    g0[3] = (unsigned int)((ga >> 32) & 0x01ffffffu)   // global_addr[56:32] -> bits 120:96
          | (2u << 30);                                // type = 2 ("image")
    int8v g1;
    g1[0] = (int)(1u << 16);            // workgroup_mask=0, data_size=1 (2 bytes)
    g1[1] = (int)(8192u << 16);         // tensor_dim0[15:0] = 8192 (bits 63:48)
    g1[2] = (int)(1u << 16);            // tensor_dim0[31:16]=0, tensor_dim1[15:0]=1
    g1[3] = (int)(8192u << 16);         // tensor_dim1[31:16]=0, tile_dim0=8192 (bits 127:112)
    g1[4] = 1;                          // tile_dim1=1, tile_dim2=0
    g1[5] = 8192;                       // tensor_dim0_stride[31:0]
    g1[6] = 0;                          // tensor_dim0_stride[47:32]=0, tensor_dim1_stride lo=0
    g1[7] = 0;
    int4v gz4 = {0, 0, 0, 0};           // groups 2/3 unused (tile dims 2-4 = 0)
    int8v gz8 = {0, 0, 0, 0, 0, 0, 0, 0};
    __builtin_amdgcn_tensor_load_to_lds(g0, g1, gz4, gz4, gz8, 0);
    __builtin_amdgcn_s_wait_tensorcnt(0);
  }
  __syncthreads();

  floatx8 c;
  const float bc = bout[v0 + r];
#pragma unroll
  for (int i = 0; i < 8; ++i) c[i] = bc;

  for (int kb = 0; kb < HID; kb += 32) {
    if (kb + 32 < HID)
      __builtin_prefetch(Wout + (size_t)(v0 + r) * HID + kb + 32, 0, 0);
    bf16x16 a = load_a_frag(h_s, HID, kb, lane);   // A from LDS (ds_load)
    bf16x16 b = load_b_frag(Wout, HID, v0, kb, lane);
    c = wmma_bf16(a, b, c);
  }

  const size_t tbase = (size_t)t * VOCAB + v0 + r;
  float mv[8]; int mi[8];
#pragma unroll
  for (int i = 0; i < 8; ++i) {
    const int row = i + 8 * half;
    out[(size_t)row * TT * VOCAB + tbase] = c[i];
    mv[i] = c[i];
    mi[i] = v0 + r;
  }
  // reduce across the 16 columns held by this half (first-occurrence tie-break)
  for (int off = 1; off < 16; off <<= 1) {
#pragma unroll
    for (int i = 0; i < 8; ++i) {
      float ov = __shfl_xor(mv[i], off, 16);
      int   oi = __shfl_xor(mi[i], off, 16);
      if (ov > mv[i] || (ov == mv[i] && oi < mi[i])) { mv[i] = ov; mi[i] = oi; }
    }
  }
  if (r == 0) {
#pragma unroll
    for (int i = 0; i < 8; ++i) {
      const int row = i + 8 * half;
      pv[(size_t)tile * BB + row] = mv[i];
      pi[(size_t)tile * BB + row] = mi[i];
    }
  }
}

__global__ __launch_bounds__(256) void argmax_reduce_kernel(
    const float* __restrict__ pv, const int* __restrict__ pi, int* __restrict__ tok) {
  const int row = blockIdx.x;  // 0..15
  const int tid = threadIdx.x;
  float bv = -3.4e38f; int bi = 0x7fffffff;
  for (int i = tid; i < VOCAB / 16; i += 256) {
    float v = pv[(size_t)i * BB + row];
    int  ix = pi[(size_t)i * BB + row];
    if (v > bv || (v == bv && ix < bi)) { bv = v; bi = ix; }
  }
  __shared__ float sv[256];
  __shared__ int   si[256];
  sv[tid] = bv; si[tid] = bi;
  __syncthreads();
  for (int s = 128; s > 0; s >>= 1) {
    if (tid < s) {
      float v = sv[tid + s]; int ix = si[tid + s];
      if (v > sv[tid] || (v == sv[tid] && ix < si[tid])) { sv[tid] = v; si[tid] = ix; }
    }
    __syncthreads();
  }
  if (tid == 0) tok[row] = si[0];
}

// ---------------------------------------------------------------- launch
extern "C" void kernel_launch(void* const* d_in, const int* in_sizes, int n_in,
                              void* d_out, int out_size, void* d_ws, size_t ws_size,
                              hipStream_t stream) {
  const int*   x        = (const int*)  d_in[0];
  const float* emb_enc  = (const float*)d_in[1];
  const float* enc_Wih  = (const float*)d_in[2];
  const float* enc_Whh  = (const float*)d_in[3];
  const float* enc_bih  = (const float*)d_in[4];
  const float* enc_bhh  = (const float*)d_in[5];
  const float* fc_enc_W = (const float*)d_in[6];
  const float* fc_enc_b = (const float*)d_in[7];
  const float* fc_dec_W = (const float*)d_in[8];
  const float* fc_dec_b = (const float*)d_in[9];
  const float* emb_dec  = (const float*)d_in[10];
  const float* dec_Wih  = (const float*)d_in[11];
  const float* dec_Whh  = (const float*)d_in[12];
  const float* dec_bih  = (const float*)d_in[13];
  const float* dec_bhh  = (const float*)d_in[14];
  const float* fc_out_W = (const float*)d_in[15];
  const float* fc_out_b = (const float*)d_in[16];
  float* out = (float*)d_out;

  char* w = (char*)d_ws;
  size_t off = 0;
  auto alloc = [&](size_t bytes) -> void* {
    void* p = w + off;
    off = (off + bytes + 255) & ~(size_t)255;
    return p;
  };
  __bf16* bWih_e = (__bf16*)alloc((size_t)3 * HID * EMB * 2);
  __bf16* bWhh_e = (__bf16*)alloc((size_t)3 * HID * HID * 2);
  __bf16* bWih_d = (__bf16*)alloc((size_t)3 * HID * EMB * 2);
  __bf16* bWhh_d = (__bf16*)alloc((size_t)3 * HID * HID * 2);
  __bf16* bWout  = (__bf16*)alloc((size_t)VOCAB * HID * 2);
  __bf16* e_seq  = (__bf16*)alloc((size_t)TT * BB * EMB * 2);
  float*  h_enc  = (float*) alloc((size_t)BB * HID * 4);
  float*  zbuf   = (float*) alloc((size_t)BB * LAT * 4);
  float*  h_f    = (float*) alloc((size_t)BB * HID * 4);
  __bf16* h_b    = (__bf16*)alloc((size_t)BB * HID * 2);
  float*  pv     = (float*) alloc((size_t)(VOCAB / 16) * BB * 4);
  int*    pi     = (int*)   alloc((size_t)(VOCAB / 16) * BB * 4);
  int*    tok    = (int*)   alloc((size_t)BB * 4);

  auto cvt = [&](const float* s, __bf16* d, int n) {
    f32_to_bf16_kernel<<<(n + 255) / 256, 256, 0, stream>>>(s, d, n);
  };
  cvt(enc_Wih, bWih_e, 3 * HID * EMB);
  cvt(enc_Whh, bWhh_e, 3 * HID * HID);
  cvt(dec_Wih, bWih_d, 3 * HID * EMB);
  cvt(dec_Whh, bWhh_d, 3 * HID * HID);
  cvt(fc_out_W, bWout, VOCAB * HID);

  gather_embed_kernel<<<(TT * BB * EMB + 255) / 256, 256, 0, stream>>>(x, emb_enc, e_seq);
  zero_t0_kernel<<<(BB * VOCAB + 255) / 256, 256, 0, stream>>>(out);

  enc_gru_kernel<<<1, 1024, 0, stream>>>(e_seq, bWih_e, bWhh_e, enc_bih, enc_bhh, h_enc);
  fc_enc_kernel<<<(BB * LAT + 255) / 256, 256, 0, stream>>>(
      h_enc, fc_enc_W, fc_enc_b, zbuf, out + (size_t)BB * TT * VOCAB);
  fc_dec_kernel<<<(BB * HID + 255) / 256, 256, 0, stream>>>(zbuf, fc_dec_W, fc_dec_b, h_f, h_b);
  init_tok_kernel<<<1, 32, 0, stream>>>(x, tok);

  for (int t = 1; t < TT; ++t) {
    dec_gru_kernel<<<1, 1024, 0, stream>>>(tok, emb_dec, bWih_d, bWhh_d, dec_bih, dec_bhh, h_f, h_b);
    logits_kernel<<<VOCAB / 16 / 8, 256, 0, stream>>>(h_b, bWout, fc_out_b, out, pv, pi, t);
    argmax_reduce_kernel<<<BB, 256, 0, stream>>>(pv, pi, tok);
  }
  (void)in_sizes; (void)n_in; (void)out_size; (void)ws_size;
}
